// Convolution_1228360646680
// MI455X (gfx1250) — matrix operand order repaired and checked by reading, not compiled, not run on hardware
//
#include <hip/hip_runtime.h>

#define N_NODES 25000
#define N_PAD   25008   // N rounded up to 16 (workspace row padding)
#define N_EDGES 400000

static_assert(N_EDGES % 128 == 0, "edge kernel assumes full blocks");
static_assert(N_PAD % 16 == 0 && N_PAD >= N_NODES, "pad");

typedef __attribute__((ext_vector_type(16))) _Float16 v16h;
typedef __attribute__((ext_vector_type(8)))  float    v8f;

#define WMMA_F16(a, b, c) \
  __builtin_amdgcn_wmma_f32_16x16x32_f16(false, (a), false, (b), (short)0, (c), false, false)

// ---- WMMA fragment helpers (CDNA5 ISA 7.12.2 layouts, wave32) ----
// Fragments staged in LDS lane-major: frag[lane*16 + t] -> one 32B LDS read/lane.

// A 16x32 f16: lane<16: row=lane, t0..7=K(klo..klo+7), t8..15=K(16+klo..); klo=8*(lane>>4)
__device__ inline void store_a_elem(_Float16* frag, int row, int k, _Float16 v) {
  const int lane = row + (((k >> 3) & 1) << 4);
  const int idx  = (k & 7) + ((k >> 4) << 3);
  frag[lane * 16 + idx] = v;
}

// B 32x16 f16: lane = col + 16*((k>>3)&1); t = (k&7) + 8*(k>>4)
__device__ inline void store_b_elem(_Float16* frag, int k, int col, _Float16 v) {
  const int lane = col + (((k >> 3) & 1) << 4);
  frag[lane * 16 + (k & 7) + ((k >> 4) << 3)] = v;
}

__device__ inline v16h load_frag(const _Float16* frag) {
  return *(const v16h*)(frag + (threadIdx.x & 31) * 16);
}

// D/C layout: col = lane&15, rows (lane>>4)*8 + r in c[r].

// ---------------------------------------------------------------------------
// Kernel 0: zero the (padded) segment-sum accumulator
// ---------------------------------------------------------------------------
__global__ void zero_kernel(float4* __restrict__ p, int n4) {
  int i = blockIdx.x * blockDim.x + threadIdx.x;
  if (i < n4) p[i] = make_float4(0.f, 0.f, 0.f, 0.f);
}

// ---------------------------------------------------------------------------
// Kernel 1: node pre-mix  xs1 = (xs*a)@w1_s*inv1 ; xv1_i = (xv_i*a)@w1_v*inv1
// One wave per 16-node tile; 8 WMMAs per tile. Stores to padded workspace:
// no guards anywhere (reads clamped to row N-1, padded rows never consumed).
// ---------------------------------------------------------------------------
__global__ __launch_bounds__(32) void node_pre_kernel(
    const float* __restrict__ node_input, const float* __restrict__ node_attr_in,
    const float* __restrict__ w1_s, const float* __restrict__ w1_v,
    float* __restrict__ xs1, float* __restrict__ xv1) {
  __shared__ __align__(32) _Float16 sWsf[2][512];
  __shared__ __align__(32) _Float16 sWvf[2][512];
  __shared__ __align__(32) _Float16 sAf[4][512];  // [0]=scalar, [1..3]=vector comps

  const int tid = threadIdx.x;
  const float inv1 = 0.17677669529663687f;  // 1/sqrt(32)

  for (int idx = tid; idx < 1024; idx += 32) {
    const int k = idx >> 5, w = idx & 31;
    store_b_elem(sWsf[w >> 4], k, w & 15, (_Float16)(w1_s[idx] * inv1));
    store_b_elem(sWvf[w >> 4], k, w & 15, (_Float16)(w1_v[idx] * inv1));
  }
  const int tile = blockIdx.x * 16;
  for (int idx = tid; idx < 512; idx += 32) {
    const int r = idx >> 5, u = idx & 31;
    const int n = min(tile + r, N_NODES - 1);  // clamped read
    const float a = node_attr_in[n];
    const float* row = node_input + (size_t)n * 128;
    store_a_elem(sAf[0], r, u, (_Float16)(row[u] * a));
    store_a_elem(sAf[1], r, u, (_Float16)(row[32 + u * 3 + 0] * a));
    store_a_elem(sAf[2], r, u, (_Float16)(row[32 + u * 3 + 1] * a));
    store_a_elem(sAf[3], r, u, (_Float16)(row[32 + u * 3 + 2] * a));
  }
  __syncthreads();

  const int lane = tid & 31;
  const int col = lane & 15;
  const int rbase = (lane >> 4) * 8;

  const v16h bs0 = load_frag(sWsf[0]), bs1 = load_frag(sWsf[1]);
  const v16h bv0 = load_frag(sWvf[0]), bv1 = load_frag(sWvf[1]);

  {  // scalar channel
    v16h a = load_frag(sAf[0]);
    v8f c0 = {}, c1 = {};
    c0 = WMMA_F16(a, bs0, c0);
    c1 = WMMA_F16(a, bs1, c1);
#pragma unroll
    for (int r = 0; r < 8; ++r) {
      const int n = tile + rbase + r;
      xs1[(size_t)n * 32 + col] = c0[r];
      xs1[(size_t)n * 32 + 16 + col] = c1[r];
    }
  }
#pragma unroll
  for (int i = 0; i < 3; ++i) {  // vector comps
    v16h a = load_frag(sAf[1 + i]);
    v8f c0 = {}, c1 = {};
    c0 = WMMA_F16(a, bv0, c0);
    c1 = WMMA_F16(a, bv1, c1);
#pragma unroll
    for (int r = 0; r < 8; ++r) {
      const int n = tile + rbase + r;
      xv1[(size_t)n * 96 + col * 3 + i] = c0[r];
      xv1[(size_t)n * 96 + (16 + col) * 3 + i] = c1[r];
    }
  }
}

// ---------------------------------------------------------------------------
// Kernel 2: edge MLP (WMMA) fused with message build + atomic segment-sum.
// 8 waves / block, each wave owns 16 edges; E % 128 == 0 -> no guards.
// ---------------------------------------------------------------------------
__global__ __launch_bounds__(256) void edge_kernel(
    const int* __restrict__ edge_src, const int* __restrict__ edge_dst,
    const float* __restrict__ edge_attr, const float* __restrict__ esc,
    const float* __restrict__ Wfc1, const float* __restrict__ Wfc2,
    const float* __restrict__ xs1, const float* __restrict__ xv1,
    float* __restrict__ mid) {
  __shared__ __align__(32) _Float16 sW1f[4][512];      // Wfc1 padded K8->32, 1/sqrt(8) folded
  __shared__ __align__(32) _Float16 sW2f[8][2][512];   // Wfc2 [ntile][khalf], 1/sqrt(64) folded
  __shared__ __align__(32) _Float16 sSCf[8][512];      // per-wave scalar-attr A frag
  __shared__ __align__(32) _Float16 sHf[8][2][512];    // per-wave hidden A frags (post-silu)

  const int tid = threadIdx.x;
  const float invs8 = 0.35355339059327373f;  // 1/sqrt(8)
  const float invs64 = 0.125f;               // 1/sqrt(64)

  for (int idx = tid; idx < 2048; idx += 256) {
    const int k = idx >> 6, n = idx & 63;
    const _Float16 v = (_Float16)((k < 8) ? Wfc1[k * 64 + n] * invs8 : 0.f);
    store_b_elem(sW1f[n >> 4], k, n & 15, v);
  }
  for (int idx = tid; idx < 8192; idx += 256) {
    const int k = idx >> 7, n = idx & 127;
    store_b_elem(sW2f[n >> 4][k >> 5], k & 31, n & 15, (_Float16)(Wfc2[idx] * invs64));
  }

  const int wid = tid >> 5;
  const int lane = tid & 31;
  const int e0 = (blockIdx.x * 8 + wid) * 16;

  if (lane == 0) {  // gfx1250 global_prefetch_b8
    __builtin_prefetch(edge_attr + (size_t)e0 * 4, 0, 1);
    __builtin_prefetch(esc + (size_t)e0 * 8, 0, 1);
  }

  for (int idx = lane; idx < 512; idx += 32) {
    const int r = idx >> 5, k = idx & 31;
    const float v = (k < 8) ? esc[(size_t)(e0 + r) * 8 + k] : 0.f;
    store_a_elem(sSCf[wid], r, k, (_Float16)v);
  }
  __syncthreads();

  const int col = lane & 15;
  const int rbase = (lane >> 4) * 8;

  // GEMM1: (16x32 padded) @ (32x64) -> h, silu -> LDS A-fragments
  {
    v16h a = load_frag(sSCf[wid]);
#pragma unroll
    for (int nt = 0; nt < 4; ++nt) {
      v16h b = load_frag(sW1f[nt]);
      v8f c = {};
      c = WMMA_F16(a, b, c);
      const int kh = nt >> 1;  // hidden k = nt*16+col, khalf = k>>5
#pragma unroll
      for (int r = 0; r < 8; ++r) {
        const float x = c[r];
        const float s = x / (1.f + __expf(-x));
        store_a_elem(sHf[wid][kh], rbase + r, (nt * 16 + col) & 31, (_Float16)s);
      }
    }
  }
  __syncthreads();

  // GEMM2: (16x64) @ (64x128); all 8 accumulators kept in VGPRs
  const v16h a0 = load_frag(sHf[wid][0]);
  const v16h a1 = load_frag(sHf[wid][1]);
  v8f acc[8];
#pragma unroll
  for (int nt = 0; nt < 8; ++nt) {
    v16h b0 = load_frag(sW2f[nt][0]);
    v16h b1 = load_frag(sW2f[nt][1]);
    v8f c = {};
    c = WMMA_F16(a0, b0, c);
    c = WMMA_F16(a1, b1, c);
    acc[nt] = c;
  }

  // Message build + scatter: one pass per edge-row, all channel groups at once.
  // Lane handles channels u0=col and u1=col+16 of each group j.
  const float kS = 0.25f;                 // 1/sqrt(NUM_NEIGH)
  const float kD = 0.14433756729740643f;  // 0.25/sqrt(3)
#pragma unroll
  for (int r = 0; r < 8; ++r) {
    const int e = e0 + rbase + r;
    const int src = edge_src[e];
    const int dst = edge_dst[e];
    const float4 ea = *(const float4*)(edge_attr + (size_t)e * 4);
    const float es0 = xs1[(size_t)src * 32 + col];
    const float es1 = xs1[(size_t)src * 32 + 16 + col];
    const float* evp = xv1 + (size_t)src * 96;
    const float ev00 = evp[col * 3 + 0], ev01 = evp[col * 3 + 1], ev02 = evp[col * 3 + 2];
    const float ev10 = evp[(16 + col) * 3 + 0], ev11 = evp[(16 + col) * 3 + 1], ev12 = evp[(16 + col) * 3 + 2];
    float* mrow = mid + (size_t)dst * 256;

    // j=0: m0 = w0 * es * ys          -> [0,32)
    unsafeAtomicAdd(mrow + col,      acc[0][r] * kS * es0 * ea.x);
    unsafeAtomicAdd(mrow + 16 + col, acc[1][r] * kS * es1 * ea.x);
    // j=3: m3 = w3 * (ev.yv)/sqrt(3)  -> [32,64)
    const float dot0 = ev00 * ea.y + ev01 * ea.z + ev02 * ea.w;
    const float dot1 = ev10 * ea.y + ev11 * ea.z + ev12 * ea.w;
    unsafeAtomicAdd(mrow + 32 + col, acc[6][r] * kD * dot0);
    unsafeAtomicAdd(mrow + 48 + col, acc[7][r] * kD * dot1);
    // j=1: m1 = w1 * es * yv_i        -> [64,160)
    const float t10 = acc[2][r] * kS * es0;
    const float t11 = acc[3][r] * kS * es1;
    unsafeAtomicAdd(mrow + 64 + col * 3 + 0, t10 * ea.y);
    unsafeAtomicAdd(mrow + 64 + col * 3 + 1, t10 * ea.z);
    unsafeAtomicAdd(mrow + 64 + col * 3 + 2, t10 * ea.w);
    unsafeAtomicAdd(mrow + 64 + (16 + col) * 3 + 0, t11 * ea.y);
    unsafeAtomicAdd(mrow + 64 + (16 + col) * 3 + 1, t11 * ea.z);
    unsafeAtomicAdd(mrow + 64 + (16 + col) * 3 + 2, t11 * ea.w);
    // j=2: m2 = w2 * ev_i * ys        -> [160,256)
    const float t20 = acc[4][r] * kS * ea.x;
    const float t21 = acc[5][r] * kS * ea.x;
    unsafeAtomicAdd(mrow + 160 + col * 3 + 0, t20 * ev00);
    unsafeAtomicAdd(mrow + 160 + col * 3 + 1, t20 * ev01);
    unsafeAtomicAdd(mrow + 160 + col * 3 + 2, t20 * ev02);
    unsafeAtomicAdd(mrow + 160 + (16 + col) * 3 + 0, t21 * ev10);
    unsafeAtomicAdd(mrow + 160 + (16 + col) * 3 + 1, t21 * ev11);
    unsafeAtomicAdd(mrow + 160 + (16 + col) * 3 + 2, t21 * ev12);
  }
}

// ---------------------------------------------------------------------------
// Kernel 3: node post-mix. K=64 concat: [s0|s1] @ [w2_s0; w2_s1], etc.
// One wave per 16-node tile; 16 WMMAs per tile. mid is padded+zeroed, so
// staging is guard-free; only the d_out store loop needs a tail guard.
// ---------------------------------------------------------------------------
__global__ __launch_bounds__(32) void node_out_kernel(
    const float* __restrict__ mid, const float* __restrict__ node_attr_out,
    const float* __restrict__ w2_s, const float* __restrict__ w2_v,
    float* __restrict__ out) {
  __shared__ __align__(32) _Float16 sWsf[2][2][512];  // [ntile][khalf]
  __shared__ __align__(32) _Float16 sWvf[2][2][512];
  __shared__ __align__(32) _Float16 sAf[4][2][512];   // [mat][khalf]

  const int tid = threadIdx.x;
  const float inv2 = 0.125f;  // 1/sqrt(64)

  for (int idx = tid; idx < 2048; idx += 32) {
    const int k = idx >> 5, w = idx & 31;   // (2,32,32) flat == row-concat 64x32
    store_b_elem(sWsf[w >> 4][k >> 5], k & 31, w & 15, (_Float16)(w2_s[idx] * inv2));
    store_b_elem(sWvf[w >> 4][k >> 5], k & 31, w & 15, (_Float16)(w2_v[idx] * inv2));
  }
  const int tile = blockIdx.x * 16;
  for (int idx = tid; idx < 1024; idx += 32) {
    const int r = idx >> 6, k = idx & 63;
    const float* m = mid + (size_t)(tile + r) * 256;  // padded rows are zero
    const int base = (k < 32) ? (64 + k * 3) : (160 + (k - 32) * 3);
    store_a_elem(sAf[0][k >> 5], r, k & 31, (_Float16)m[k]);
    store_a_elem(sAf[1][k >> 5], r, k & 31, (_Float16)m[base + 0]);
    store_a_elem(sAf[2][k >> 5], r, k & 31, (_Float16)m[base + 1]);
    store_a_elem(sAf[3][k >> 5], r, k & 31, (_Float16)m[base + 2]);
  }
  __syncthreads();

  const int lane = tid & 31;
  const int col = lane & 15;
  const int rbase = (lane >> 4) * 8;
  const bool full = (tile + 16 <= N_NODES);

  float bmul[8];
#pragma unroll
  for (int r = 0; r < 8; ++r)
    bmul[r] = node_attr_out[min(tile + rbase + r, N_NODES - 1)];

  v16h wb[2][2][2];  // [s/v][ntile][khalf]
#pragma unroll
  for (int nt = 0; nt < 2; ++nt)
#pragma unroll
    for (int kh = 0; kh < 2; ++kh) {
      wb[0][nt][kh] = load_frag(sWsf[nt][kh]);
      wb[1][nt][kh] = load_frag(sWvf[nt][kh]);
    }

#pragma unroll
  for (int m = 0; m < 4; ++m) {
    const int wsel = (m == 0) ? 0 : 1;
    v16h a0 = load_frag(sAf[m][0]);
    v16h a1 = load_frag(sAf[m][1]);
#pragma unroll
    for (int nt = 0; nt < 2; ++nt) {
      v8f c = {};
      c = WMMA_F16(a0, wb[wsel][nt][0], c);
      c = WMMA_F16(a1, wb[wsel][nt][1], c);
      const int w = nt * 16 + col;
      const int oi = (m == 0) ? w : (32 + w * 3 + (m - 1));
      if (full) {
#pragma unroll
        for (int r = 0; r < 8; ++r) {
          const int n = tile + rbase + r;
          out[(size_t)n * 128 + oi] = c[r] * bmul[r];
        }
      } else {
#pragma unroll
        for (int r = 0; r < 8; ++r) {
          const int n = tile + rbase + r;
          if (n < N_NODES) out[(size_t)n * 128 + oi] = c[r] * bmul[r];
        }
      }
    }
  }
}

// ---------------------------------------------------------------------------
extern "C" void kernel_launch(void* const* d_in, const int* in_sizes, int n_in,
                              void* d_out, int out_size, void* d_ws, size_t ws_size,
                              hipStream_t stream) {
  const float* node_input = (const float*)d_in[0];
  const float* nai        = (const float*)d_in[1];
  const float* nao        = (const float*)d_in[2];
  const int*   esrc       = (const int*)d_in[3];
  const int*   edst       = (const int*)d_in[4];
  const float* eattr      = (const float*)d_in[5];
  const float* escal      = (const float*)d_in[6];
  const float* w1_s       = (const float*)d_in[7];
  const float* w1_v       = (const float*)d_in[8];
  const float* Wfc1       = (const float*)d_in[9];
  const float* Wfc2       = (const float*)d_in[10];
  const float* w2_s       = (const float*)d_in[11];
  const float* w2_v       = (const float*)d_in[12];
  float* out = (float*)d_out;

  float* xs1 = (float*)d_ws;                // N_PAD x 32
  float* xv1 = xs1 + (size_t)N_PAD * 32;    // N_PAD x 96 (u-major, xyz interleaved)
  float* mid = xv1 + (size_t)N_PAD * 96;    // N_PAD x 256 (segment-sum accumulator)

  const int nmid4 = N_PAD * 256 / 4;
  zero_kernel<<<(nmid4 + 255) / 256, 256, 0, stream>>>((float4*)mid, nmid4);
  node_pre_kernel<<<N_PAD / 16, 32, 0, stream>>>(node_input, nai, w1_s, w1_v, xs1, xv1);
  edge_kernel<<<N_EDGES / 128, 256, 0, stream>>>(esrc, edst, eattr, escal,
                                                 Wfc1, Wfc2, xs1, xv1, mid);
  node_out_kernel<<<N_PAD / 16, 32, 0, stream>>>(mid, nao, w2_s, w2_v, out);
}
